// HSTv7Agile_46291157516427
// MI455X (gfx1250) — matrix operand order repaired and checked by reading, not compiled, not run on hardware
//
#include <hip/hip_runtime.h>
#include <hip/hip_bf16.h>

// ---------------------------------------------------------------------------
// Problem constants
// ---------------------------------------------------------------------------
#define LL 6
#define DD 1024
#define HH 16
#define FF 4096
#define BB 4
#define SS 2048
#define HDIM 64
#define MDIM (BB * SS)          // 8192 rows
#define MD ((size_t)MDIM * DD)  // 8,388,608 elements

typedef __bf16 bf16;
typedef unsigned int u32;
typedef __attribute__((ext_vector_type(16))) __bf16 v16bf;
typedef __attribute__((ext_vector_type(8)))  float  v8f;
typedef __attribute__((ext_vector_type(4)))  unsigned int v4u;
typedef __attribute__((ext_vector_type(8)))  int v8i;
typedef __attribute__((ext_vector_type(4)))  int v4i;

union AFrag { v16bf v; u32 u[8]; };

#if __has_builtin(__builtin_amdgcn_tensor_load_to_lds) && \
    __has_builtin(__builtin_amdgcn_s_wait_tensorcnt)
#define USE_TDM 1
#else
#define USE_TDM 0
#endif

// ---------------------------------------------------------------------------
// LayerNorm: f32 row (D=1024) -> bf16 row. One block (256 thr) per row.
// ---------------------------------------------------------------------------
__global__ __launch_bounds__(256)
void layernorm_bf16_k(const float* __restrict__ x, const float* __restrict__ g,
                      const float* __restrict__ b, bf16* __restrict__ out)
{
    __shared__ float red[2][8];
    const int row = blockIdx.x;
    const float* xr = x + (size_t)row * DD;
    float vals[4];
    float s = 0.f, s2 = 0.f;
    #pragma unroll
    for (int i = 0; i < 4; i++) {
        float v = xr[threadIdx.x + i * 256];
        vals[i] = v; s += v; s2 += v * v;
    }
    #pragma unroll
    for (int o = 16; o > 0; o >>= 1) { s += __shfl_xor(s, o, 32); s2 += __shfl_xor(s2, o, 32); }
    const int w = threadIdx.x >> 5, ln = threadIdx.x & 31;
    if (ln == 0) { red[0][w] = s; red[1][w] = s2; }
    __syncthreads();
    if (w == 0) {
        s  = (ln < 8) ? red[0][ln] : 0.f;
        s2 = (ln < 8) ? red[1][ln] : 0.f;
        #pragma unroll
        for (int o = 4; o > 0; o >>= 1) { s += __shfl_xor(s, o, 32); s2 += __shfl_xor(s2, o, 32); }
        if (ln == 0) { red[0][0] = s; red[1][0] = s2; }
    }
    __syncthreads();
    const float mean = red[0][0] * (1.f / DD);
    const float var  = red[1][0] * (1.f / DD) - mean * mean;
    const float rstd = rsqrtf(var + 1e-5f);
    bf16* orow = out + (size_t)row * DD;
    #pragma unroll
    for (int i = 0; i < 4; i++) {
        int c = threadIdx.x + i * 256;
        orow[c] = (bf16)((vals[i] - mean) * rstd * g[c] + b[c]);
    }
}

// ---------------------------------------------------------------------------
// Weight transpose + convert: W (KxN f32 row-major) -> Wt (NxK bf16 row-major)
// grid (N/32, K/32), block 256 (= 32x8)
// ---------------------------------------------------------------------------
__global__ __launch_bounds__(256)
void transpose_to_bf16_k(const float* __restrict__ W, bf16* __restrict__ Wt,
                         int K, int N)
{
    __shared__ float t[32][33];
    const int n0 = blockIdx.x * 32, k0 = blockIdx.y * 32;
    const int tx = threadIdx.x & 31, ty = threadIdx.x >> 5;
    #pragma unroll
    for (int r = 0; r < 4; r++)
        t[ty * 4 + r][tx] = W[(size_t)(k0 + ty * 4 + r) * N + n0 + tx];
    __syncthreads();
    #pragma unroll
    for (int r = 0; r < 4; r++)
        Wt[(size_t)(n0 + ty * 4 + r) * K + k0 + tx] = (bf16)t[tx][ty * 4 + r];
}

// ---------------------------------------------------------------------------
// TDM descriptor: 2-D tile (tile_dim0=32 elems along K, tile_dim1=128 rows)
// from a row-major (rows x K) bf16 tensor into LDS with HW row padding:
// pad_interval=3 (pad after 16 DWORDs = 64B = one row), pad_amount=3 (4 DWORDs
// = 16B) -> LDS row stride 80B = 40 bf16 (matches LDT below).
// ---------------------------------------------------------------------------
#if USE_TDM
__device__ __forceinline__ void tdm_load_tile(const bf16* gsrc, unsigned lds_byte,
                                              int dimK, int dimRows, int strideK)
{
    const unsigned long long ga = (unsigned long long)gsrc;
    v4u g0;
    g0[0] = 1u;                                             // count=1 (valid D#)
    g0[1] = lds_byte;                                       // lds_addr (bytes)
    g0[2] = (u32)ga;                                        // global_addr[31:0]
    g0[3] = (u32)((ga >> 32) & 0x1FFFFFFu) | (2u << 30);    // addr[56:32] | type=2
    v8i g1;
    g1[0] = (int)((1u << 16)        // data_size = 1 -> 2 bytes
                | (1u << 20)        // pad_enable
                | (3u << 22)        // pad_interval: 16 DWORDs
                | (3u << 25));      // pad_amount:   4 DWORDs
    g1[1] = (int)(((u32)dimK & 0xFFFFu) << 16);             // tensor_dim0[15:0]
    g1[2] = (int)((((u32)dimK >> 16) & 0xFFFFu)
                | (((u32)dimRows & 0xFFFFu) << 16));        // dim0 hi | dim1 lo
    g1[3] = (int)((((u32)dimRows >> 16) & 0xFFFFu)
                | (32u << 16));                             // dim1 hi | tile_dim0=32
    g1[4] = 128;                                            // tile_dim1=128, tile_dim2=0
    g1[5] = strideK;                                        // tensor_dim0_stride[31:0]
    g1[6] = 0;                                              // stride0 hi | stride1 lo
    g1[7] = 0;
    v4i z4 = {};
    v8i z8 = {};
    // 6-arg toolchain variant: (g0, g1, g2, g3, g4, cpol)
    __builtin_amdgcn_tensor_load_to_lds(g0, g1, z4, z4, z8, 0);
}
#endif

// ---------------------------------------------------------------------------
// GEMM: out = epilogue( A(MxK bf16) @ Wt(NxK bf16)^T + bias [+ resid] )
// Tile 128x128x32, 8 waves (4x2), wave tile 32x64 = 2x4 wmma 16x16 tiles.
// Tiles arrive via TDM (tensor_load_to_lds, double-buffered, TENSORcnt-gated).
// ---------------------------------------------------------------------------
template<bool RELU, bool BF16OUT, bool RESID>
__global__ __launch_bounds__(256)
void gemm_bf16_k(const bf16* __restrict__ A, const bf16* __restrict__ Wt,
                 const float* __restrict__ bias, const float* __restrict__ resid,
                 float* __restrict__ outF, bf16* __restrict__ outB,
                 int M, int N, int K)
{
    constexpr int TM = 128, TN = 128, TK = 32;
    constexpr int LDT = TK + 8;                    // 40 bf16 = 80B padded rows
    constexpr unsigned TILE_B = TM * LDT * 2;      // 10240 B per tile
    constexpr unsigned BUF_B  = 2 * TILE_B;        // A+B tile per buffer
    extern __shared__ char smem[];                 // 2 * BUF_B = 40 KB dynamic

    const int m0 = blockIdx.y * TM, n0 = blockIdx.x * TN;
    const int tid  = threadIdx.x;
    const int wid  = tid >> 5, lane = tid & 31;
    const int half = lane >> 4, ml = lane & 15;
    const int wm = wid >> 1, wn = wid & 1;

    v8f acc[2][4];
    #pragma unroll
    for (int i = 0; i < 2; i++)
        #pragma unroll
        for (int j = 0; j < 4; j++)
            acc[i][j] = (v8f){};

#if USE_TDM
    const unsigned ldsBase = (unsigned)__builtin_amdgcn_groupstaticsize();
    if (wid == 0) {   // one wave drives the DMA engine
        tdm_load_tile(A  + (size_t)m0 * K, ldsBase,          K, M, K);
        tdm_load_tile(Wt + (size_t)n0 * K, ldsBase + TILE_B, K, N, K);
    }
#endif

    int cur = 0;
    for (int k0 = 0; k0 < K; k0 += TK) {
        __syncthreads();   // everyone done reading the buffer we refill next
#if USE_TDM
        if (wid == 0) {
            if (k0 + TK < K) {
                const unsigned nb = ldsBase + (unsigned)(cur ^ 1) * BUF_B;
                tdm_load_tile(A  + (size_t)m0 * K + k0 + TK, nb,          K, M, K);
                tdm_load_tile(Wt + (size_t)n0 * K + k0 + TK, nb + TILE_B, K, N, K);
                __builtin_amdgcn_s_wait_tensorcnt(2);  // current tile's 2 DMAs done
            } else {
                __builtin_amdgcn_s_wait_tensorcnt(0);
            }
        }
#else
        {   // fallback: synchronous VGPR staging
            bf16* dA = (bf16*)(smem + (size_t)cur * BUF_B);
            bf16* dB = (bf16*)(smem + (size_t)cur * BUF_B + TILE_B);
            for (int c = tid; c < (TM * TK) / 8; c += 256) {
                int row = c >> 2, kc = (c & 3) * 8;
                *(uint4*)&dA[row * LDT + kc] =
                    *(const uint4*)&A[(size_t)(m0 + row) * K + k0 + kc];
                *(uint4*)&dB[row * LDT + kc] =
                    *(const uint4*)&Wt[(size_t)(n0 + row) * K + k0 + kc];
            }
        }
#endif
        __syncthreads();

        const bf16* sA = (const bf16*)(smem + (size_t)cur * BUF_B);
        const bf16* sB = (const bf16*)(smem + (size_t)cur * BUF_B + TILE_B);

        AFrag a[2];
        #pragma unroll
        for (int i = 0; i < 2; i++) {
            const int row = wm * 32 + i * 16 + ml;
            #pragma unroll
            for (int p = 0; p < 8; p++) {
                const int kk = 2 * p + 16 * half;
                a[i].u[p] = *(const u32*)&sA[row * LDT + kk];
            }
        }
        #pragma unroll
        for (int j = 0; j < 4; j++) {
            AFrag b;
            const int col = wn * 64 + j * 16 + ml;
            #pragma unroll
            for (int p = 0; p < 8; p++)
                b.u[p] = *(const u32*)&sB[col * LDT + 2 * p + 16 * half];
            #pragma unroll
            for (int i = 0; i < 2; i++)
                acc[i][j] = __builtin_amdgcn_wmma_f32_16x16x32_bf16(
                    false, a[i].v, false, b.v, (short)0, acc[i][j], false, false);
        }
        cur ^= 1;
    }

    // Epilogue. D layout: lane n = N col, VGPR r -> M = r + 8*half.
    #pragma unroll
    for (int i = 0; i < 2; i++)
        #pragma unroll
        for (int j = 0; j < 4; j++)
            #pragma unroll
            for (int r = 0; r < 8; r++) {
                const size_t gm = (size_t)m0 + wm * 32 + i * 16 + r + 8 * half;
                const size_t gn = (size_t)n0 + wn * 64 + j * 16 + ml;
                float v = acc[i][j][r] + bias[gn];
                if (RELU) v = v > 0.f ? v : 0.f;
                if (RESID) v += resid[gm * N + gn];
                if (BF16OUT) outB[gm * N + gn] = (bf16)v;
                else         outF[gm * N + gn] = v;
            }
}

// ---------------------------------------------------------------------------
// Flash attention (causal). grid (S/64, H, B), block 128 (4 waves x 16 rows).
// Q/K/V/O are (B,S,D) bf16 row-major; head h at column offset h*64.
// ---------------------------------------------------------------------------
__global__ __launch_bounds__(128)
void flash_attn_k(const bf16* __restrict__ Q, const bf16* __restrict__ Kb,
                  const bf16* __restrict__ Vb, bf16* __restrict__ O)
{
    constexpr int KS = 72;  // sK  row stride (bf16): 144B, 16B aligned
    constexpr int VS = 40;  // sVt row stride
    constexpr int PS = 40;  // sP  row stride
    __shared__ bf16 sK[32 * KS];     // [key][feat]
    __shared__ bf16 sVt[64 * VS];    // [feat][key]
    __shared__ bf16 sP[4][16 * PS];  // per-wave P tile [query][key]

    const int q0 = blockIdx.x * 64;
    const int h  = blockIdx.y;
    const int bi = blockIdx.z;
    const int tid  = threadIdx.x;
    const int wid  = tid >> 5, lane = tid & 31;
    const int half = lane >> 4, ml = lane & 15;
    const float scale = 0.125f;  // HD^-0.5
    const size_t base = ((size_t)bi * SS) * DD + (size_t)h * HDIM;

    // Q fragments (this wave's 16 query rows), split HD=64 -> two K=32 frags
    AFrag qf[2];
    {
        const bf16* qrow = Q + base + (size_t)(q0 + wid * 16 + ml) * DD;
        #pragma unroll
        for (int t = 0; t < 2; t++)
            #pragma unroll
            for (int p = 0; p < 8; p++) {
                const int kk = t * 32 + 2 * p + 16 * half;
                qf[t].u[p] = *(const u32*)&qrow[kk];
            }
    }

    v8f out0 = (v8f){}, out1 = (v8f){}, out2 = (v8f){}, out3 = (v8f){};
    float mrow[8], lrow[8];
    #pragma unroll
    for (int r = 0; r < 8; r++) { mrow[r] = -INFINITY; lrow[r] = 0.f; }

    const int nblk = q0 / 32 + 2;  // causal: keys 0 .. q0+63
    for (int kb = 0; kb < nblk; kb++) {
        __syncthreads();
        // stage K block (32 keys x 64 feats), row-major
        for (int c = tid; c < 256; c += 128) {
            const int key = c >> 3, fc = (c & 7) * 8;
            *(uint4*)&sK[key * KS + fc] =
                *(const uint4*)&Kb[base + (size_t)(kb * 32 + key) * DD + fc];
        }
        // stage V block transposed: sVt[feat][key]
        for (int e = tid; e < 2048; e += 128) {
            const int key = e >> 6, f = e & 63;
            sVt[f * VS + key] = Vb[base + (size_t)(kb * 32 + key) * DD + f];
        }
        __syncthreads();

        // scores: two 16x16 tiles (keys j*16..), each = chain of 2 wmma over HD
        v8f sc[2];
        #pragma unroll
        for (int j = 0; j < 2; j++) {
            AFrag kf0, kf1;
            const int key = j * 16 + ml;
            #pragma unroll
            for (int p = 0; p < 8; p++) {
                kf0.u[p] = *(const u32*)&sK[key * KS +      2 * p + 16 * half];
                kf1.u[p] = *(const u32*)&sK[key * KS + 32 + 2 * p + 16 * half];
            }
            v8f s = (v8f){};
            s = __builtin_amdgcn_wmma_f32_16x16x32_bf16(false, qf[0].v, false, kf0.v, (short)0, s, false, false);
            s = __builtin_amdgcn_wmma_f32_16x16x32_bf16(false, qf[1].v, false, kf1.v, (short)0, s, false, false);
            sc[j] = s;
        }

        // online softmax, per query row (r,half); 16-lane reductions
        #pragma unroll
        for (int r = 0; r < 8; r++) {
            const int qi  = q0 + wid * 16 + r + 8 * half;
            const int k0i = kb * 32 + ml;
            const int k1i = k0i + 16;
            float a0 = (k0i <= qi) ? sc[0][r] * scale : -INFINITY;
            float a1 = (k1i <= qi) ? sc[1][r] * scale : -INFINITY;
            float mx = fmaxf(a0, a1);
            #pragma unroll
            for (int o = 8; o > 0; o >>= 1) mx = fmaxf(mx, __shfl_xor(mx, o, 16));
            const float mi    = fmaxf(mrow[r], mx);
            const float alpha = __expf(mrow[r] - mi);
            const float p0 = __expf(a0 - mi);
            const float p1 = __expf(a1 - mi);
            float ps = p0 + p1;
            #pragma unroll
            for (int o = 8; o > 0; o >>= 1) ps += __shfl_xor(ps, o, 16);
            lrow[r] = lrow[r] * alpha + ps;
            mrow[r] = mi;
            out0[r] *= alpha; out1[r] *= alpha; out2[r] *= alpha; out3[r] *= alpha;
            const int prow = r + 8 * half;
            sP[wid][prow * PS + ml]      = (bf16)p0;   // wave-private: DS in-order
            sP[wid][prow * PS + 16 + ml] = (bf16)p1;
        }

        // P (16x32) as A fragment
        AFrag pf;
        #pragma unroll
        for (int p = 0; p < 8; p++) {
            const int kk = 2 * p + 16 * half;
            pf.u[p] = *(const u32*)&sP[wid][ml * PS + kk];
        }
        // P @ V: four 16-wide feature tiles
        #pragma unroll
        for (int j2 = 0; j2 < 4; j2++) {
            AFrag vf;
            const int f = j2 * 16 + ml;
            #pragma unroll
            for (int p = 0; p < 8; p++)
                vf.u[p] = *(const u32*)&sVt[f * VS + 2 * p + 16 * half];
            v8f& o = (j2 == 0 ? out0 : j2 == 1 ? out1 : j2 == 2 ? out2 : out3);
            o = __builtin_amdgcn_wmma_f32_16x16x32_bf16(false, pf.v, false, vf.v, (short)0, o, false, false);
        }
    }

    #pragma unroll
    for (int r = 0; r < 8; r++) {
        const float inv = 1.0f / lrow[r];
        const size_t orow = base + (size_t)(q0 + wid * 16 + r + 8 * half) * DD;
        O[orow + 0  + ml] = (bf16)(out0[r] * inv);
        O[orow + 16 + ml] = (bf16)(out1[r] * inv);
        O[orow + 32 + ml] = (bf16)(out2[r] * inv);
        O[orow + 48 + ml] = (bf16)(out3[r] * inv);
    }
}

// ---------------------------------------------------------------------------
// Confidence path + control
// ---------------------------------------------------------------------------
__global__ __launch_bounds__(256)
void seq_mean_k(const float* __restrict__ x2, float* __restrict__ meanb)
{
    const int idx = blockIdx.x * 256 + threadIdx.x;  // B*D = 4096
    const int b = idx >> 10, d = idx & 1023;
    const float* p = x2 + (size_t)b * SS * DD + d;
    float s = 0.f;
    for (int t = 0; t < SS; t++) s += p[(size_t)t * DD];
    meanb[idx] = s * (1.f / SS);
}

__global__ __launch_bounds__(256)
void conf_update_k(const float* __restrict__ meanb, const float* __restrict__ cW,
                   const float* __restrict__ cb, int* __restrict__ flags, int layer)
{
    __shared__ float red[8];
    __shared__ float confsh;
    if (threadIdx.x == 0) confsh = 0.f;
    __syncthreads();
    for (int b = 0; b < BB; b++) {
        float s = 0.f;
        for (int d = threadIdx.x; d < DD; d += 256) s += meanb[b * DD + d] * cW[d];
        #pragma unroll
        for (int o = 16; o > 0; o >>= 1) s += __shfl_xor(s, o, 32);
        if ((threadIdx.x & 31) == 0) red[threadIdx.x >> 5] = s;
        __syncthreads();
        if (threadIdx.x == 0) {
            float t = 0.f;
            for (int w = 0; w < 8; w++) t += red[w];
            confsh += 1.f / (1.f + __expf(-(t + cb[0])));
        }
        __syncthreads();
    }
    if (threadIdx.x == 0) {
        const float conf = confsh * 0.25f;
        if (!flags[0] && layer >= 1 && conf > 0.93f) { flags[0] = 1; flags[1] = layer + 1; }
    }
}

__global__ __launch_bounds__(256)
void commit_k(const float* __restrict__ x2, float* __restrict__ cur,
              const int* __restrict__ flags)
{
    if (flags[0]) return;  // frozen: keep cur
    const size_t i = (size_t)blockIdx.x * 256 + threadIdx.x;
    cur[i] = x2[i];
}

__global__ __launch_bounds__(256)
void init_k(const float* __restrict__ x, float* __restrict__ cur, int* __restrict__ flags)
{
    const size_t i = (size_t)blockIdx.x * 256 + threadIdx.x;
    cur[i] = x[i];
    if (i == 0) { flags[0] = 0; flags[1] = LL; }
}

__global__ __launch_bounds__(256)
void finalize_k(const float* __restrict__ cur, const int* __restrict__ flags,
                float* __restrict__ out)
{
    const size_t i = (size_t)blockIdx.x * 256 + threadIdx.x;
    out[i] = cur[i];
    if (i == 0) out[MD] = (float)flags[1];
}

// ---------------------------------------------------------------------------
// Host launch
// ---------------------------------------------------------------------------
extern "C" void kernel_launch(void* const* d_in, const int* in_sizes, int n_in,
                              void* d_out, int out_size, void* d_ws, size_t ws_size,
                              hipStream_t stream) {
    (void)in_sizes; (void)n_in; (void)out_size; (void)ws_size;
    const float* x    = (const float*)d_in[0];
    const float* Wq   = (const float*)d_in[1];
    const float* bq   = (const float*)d_in[2];
    const float* Wk   = (const float*)d_in[3];
    const float* bk   = (const float*)d_in[4];
    const float* Wv   = (const float*)d_in[5];
    const float* bv   = (const float*)d_in[6];
    const float* Wo   = (const float*)d_in[7];
    const float* bo   = (const float*)d_in[8];
    const float* W1   = (const float*)d_in[9];
    const float* b1   = (const float*)d_in[10];
    const float* W2   = (const float*)d_in[11];
    const float* b2   = (const float*)d_in[12];
    const float* ln1g = (const float*)d_in[13];
    const float* ln1b = (const float*)d_in[14];
    const float* ln2g = (const float*)d_in[15];
    const float* ln2b = (const float*)d_in[16];
    const float* cW   = (const float*)d_in[17];
    const float* cb   = (const float*)d_in[18];
    float* out = (float*)d_out;

    // carve workspace
    char* wp = (char*)d_ws;
    auto take = [&](size_t bytes) -> void* {
        void* p = wp; wp += (bytes + 255) & ~(size_t)255; return p;
    };
    float* cur   = (float*)take(MD * 4);
    float* x1    = (float*)take(MD * 4);
    float* x2    = (float*)take(MD * 4);
    bf16*  xn    = (bf16*)take(MD * 2);
    bf16*  qb    = (bf16*)take(MD * 2);
    bf16*  kb    = (bf16*)take(MD * 2);
    bf16*  vb    = (bf16*)take(MD * 2);
    bf16*  ob    = (bf16*)take(MD * 2);
    bf16*  hb    = (bf16*)take((size_t)MDIM * FF * 2);
    bf16*  wT    = (bf16*)take((size_t)DD * FF * 2);
    float* meanb = (float*)take((size_t)BB * DD * 4);
    int*   flags = (int*)take(2 * sizeof(int));

    const dim3 blk256(256), blk128(128);
    const dim3 gDD(DD / 128, MDIM / 128);          // (8, 64)
    const dim3 gDF(FF / 128, MDIM / 128);          // (32, 64)
    const dim3 gTDD(DD / 32, DD / 32);             // (32, 32)
    const dim3 gTDF(FF / 32, DD / 32);             // (128, 32)
    const dim3 gTFD(DD / 32, FF / 32);             // (32, 128)
    const dim3 gAttn(SS / 64, HH, BB);             // (32, 16, 4)
    const int  nElemBlocks = (int)(MD / 256);      // 32768
    const size_t gemmLds = 2u * 2u * 128u * 40u * 2u;  // 40960 B dynamic LDS

    init_k<<<nElemBlocks, blk256, 0, stream>>>(x, cur, flags);

    for (int i = 0; i < LL; i++) {
        // ----- attention branch -----
        layernorm_bf16_k<<<MDIM, blk256, 0, stream>>>(cur, ln1g + i * DD, ln1b + i * DD, xn);

        transpose_to_bf16_k<<<gTDD, blk256, 0, stream>>>(Wq + (size_t)i * DD * DD, wT, DD, DD);
        gemm_bf16_k<false, true, false><<<gDD, blk256, gemmLds, stream>>>(
            xn, wT, bq + i * DD, nullptr, nullptr, qb, MDIM, DD, DD);
        transpose_to_bf16_k<<<gTDD, blk256, 0, stream>>>(Wk + (size_t)i * DD * DD, wT, DD, DD);
        gemm_bf16_k<false, true, false><<<gDD, blk256, gemmLds, stream>>>(
            xn, wT, bk + i * DD, nullptr, nullptr, kb, MDIM, DD, DD);
        transpose_to_bf16_k<<<gTDD, blk256, 0, stream>>>(Wv + (size_t)i * DD * DD, wT, DD, DD);
        gemm_bf16_k<false, true, false><<<gDD, blk256, gemmLds, stream>>>(
            xn, wT, bv + i * DD, nullptr, nullptr, vb, MDIM, DD, DD);

        flash_attn_k<<<gAttn, blk128, 0, stream>>>(qb, kb, vb, ob);

        transpose_to_bf16_k<<<gTDD, blk256, 0, stream>>>(Wo + (size_t)i * DD * DD, wT, DD, DD);
        gemm_bf16_k<false, false, true><<<gDD, blk256, gemmLds, stream>>>(
            ob, wT, bo + i * DD, cur, x1, nullptr, MDIM, DD, DD);   // x1 = cur + o@Wo + bo

        // ----- MLP branch -----
        layernorm_bf16_k<<<MDIM, blk256, 0, stream>>>(x1, ln2g + i * DD, ln2b + i * DD, xn);

        transpose_to_bf16_k<<<gTDF, blk256, 0, stream>>>(W1 + (size_t)i * DD * FF, wT, DD, FF);
        gemm_bf16_k<true, true, false><<<gDF, blk256, gemmLds, stream>>>(
            xn, wT, b1 + i * FF, nullptr, nullptr, hb, MDIM, FF, DD); // h = relu(xn@W1+b1)

        transpose_to_bf16_k<<<gTFD, blk256, 0, stream>>>(W2 + (size_t)i * FF * DD, wT, FF, DD);
        gemm_bf16_k<false, false, true><<<gDD, blk256, gemmLds, stream>>>(
            hb, wT, b2 + i * DD, x1, x2, nullptr, MDIM, DD, FF);    // x2 = x1 + h@W2 + b2

        // ----- early-exit control (commit uses PREVIOUS done; then update) -----
        commit_k<<<nElemBlocks, blk256, 0, stream>>>(x2, cur, flags);
        seq_mean_k<<<(BB * DD) / 256, blk256, 0, stream>>>(x2, meanb);
        conf_update_k<<<1, blk256, 0, stream>>>(meanb, cW + i * DD, cb + i, flags, i);
    }

    finalize_k<<<nElemBlocks, blk256, 0, stream>>>(cur, flags, out);
}
